// EfficientAttentionBlock_39668317946293
// MI455X (gfx1250) — compile-verified, matrix operand order checked by reading, not run on hardware
//
#include <hip/hip_runtime.h>

// ---------------------------------------------------------------------------
// CDNA5 / gfx1250 fused attention block:
//   LN -> (bf16) -> QKV WMMA GEMM -> flash attention (WMMA + TDM + online
//   softmax) -> proj WMMA GEMM.
// Wave32, v_wmma_f32_16x16x32_bf16 everywhere, tensor_load_to_lds for tiles.
// ---------------------------------------------------------------------------

typedef unsigned int   u32;
typedef unsigned short u16;
typedef __attribute__((ext_vector_type(16))) __bf16 v16bf;
typedef __attribute__((ext_vector_type(8)))  float  v8f;
typedef __attribute__((ext_vector_type(4)))  u32    v4u;
typedef __attribute__((ext_vector_type(8)))  int    v8i;
typedef __attribute__((ext_vector_type(4)))  int    v4i;

#define DIMC   256
#define NTOK   4096        // H*W
#define NHEAD  4
#define HDIM   64

__device__ __forceinline__ u16 f2bf(float f) {
  u32 u = __float_as_uint(f);
  u32 r = u + 0x7FFFu + ((u >> 16) & 1u);
  return (u16)(r >> 16);
}

union ABfrag { v16bf v; u32 w[8]; u16 h[16]; };

__device__ __forceinline__ v8f wmma_bf16(v16bf a, v16bf b, v8f c) {
  return __builtin_amdgcn_wmma_f32_16x16x32_bf16(false, a, false, b, (short)0, c,
                                                 false, false);
}

// K index for element-pair i of a 16-bit A/B fragment (ISA 7.12.2):
// lanes 0-15 cover K={0..7,16..23}, lanes 16-31 cover K={8..15,24..31}.
__device__ __forceinline__ int kpat(int i, int hi) {
  return (i < 4) ? (hi * 8 + 2 * i) : (16 + hi * 8 + 2 * (i - 4));
}

// ------------------------- TDM: 1-D bf16 tile -> LDS ------------------------
// D# group0/group1 per CDNA5 ISA (data_size=2B,
// 1-D tile of nelem elements, no padding/multicast).
// This toolchain exposes the 6-arg builtin:
//   (uint32x4 g0, int32x8 g1, int32x4 g2, int32x4 g3, int32x8 extra, i32 cpol)
__device__ __forceinline__ void tdm_load_bf16(u32 lds_byte_off, const void* gptr,
                                              u32 nelem) {
  unsigned long long ga = (unsigned long long)gptr;
  v4u g0;
  g0[0] = 1u;                                   // count = 1 (valid descriptor)
  g0[1] = lds_byte_off;                         // lds_addr
  g0[2] = (u32)(ga & 0xFFFFFFFFu);              // global_addr[31:0]
  g0[3] = (u32)((ga >> 32) & 0x01FFFFFFu) | (2u << 30); // addr[56:32], type=2
  v8i g1;
  g1[0] = (int)(1u << 16);                      // data_size = 1 (2 bytes)
  g1[1] = (int)((nelem & 0xFFFFu) << 16);       // tensor_dim0[15:0]
  g1[2] = (int)((nelem >> 16) & 0xFFFFu) | (1 << 16); // td0[31:16], td1=1
  g1[3] = (int)((nelem & 0xFFFFu) << 16);       // tile_dim0
  g1[4] = 0;                                    // tile_dim1 = tile_dim2 = 0
  g1[5] = (int)nelem;                           // tensor_dim0_stride[31:0]
  g1[6] = 0;
  g1[7] = 0;
  v4i z4 = {0, 0, 0, 0};
  v8i z8 = {0, 0, 0, 0, 0, 0, 0, 0};
  __builtin_amdgcn_tensor_load_to_lds(g0, g1, z4, z4, z8, 0);
}

// --------------------------- kernel 1: weights -----------------------------
__global__ __launch_bounds__(256) void convert_weights_kernel(
    const float* __restrict__ wqkv, const float* __restrict__ wproj,
    u16* __restrict__ wq_bf, u16* __restrict__ wpT_bf) {
  int idx = blockIdx.x * 256 + threadIdx.x;
  if (idx < DIMC * 3 * DIMC) wq_bf[idx] = f2bf(wqkv[idx]);
  if (idx < DIMC * DIMC) {                       // wpT[k][n] = wproj[n][k]
    int k = idx >> 8, n = idx & 255;
    wpT_bf[idx] = f2bf(wproj[n * DIMC + k]);
  }
}

// --------------------------- kernel 2: LayerNorm ---------------------------
// One block = 64 tokens. Reads x[b][c][n] coalesced along n; two-pass (the
// second read is an L2 hit: block footprint is 64KB, L2 is 192MB).
__global__ __launch_bounds__(256) void ln_bf16_kernel(
    const float* __restrict__ x, const float* __restrict__ g,
    const float* __restrict__ bta, u16* __restrict__ t) {
  __shared__ float red[2][4][64];
  __shared__ float stats[2][64];
  int blk = blockIdx.x;
  int b = blk >> 6, n0 = (blk & 63) << 6;
  int tid = threadIdx.x, tok = tid & 63, cl = tid >> 6;
  const float* xb = x + ((size_t)b * DIMC * NTOK) + n0;
  float s = 0.f, s2 = 0.f;
  for (int cc = 0; cc < 64; ++cc) {
    int c = cc * 4 + cl;
    float v = xb[(size_t)c * NTOK + tok];
    s += v; s2 += v * v;
  }
  red[0][cl][tok] = s; red[1][cl][tok] = s2;
  __syncthreads();
  if (tid < 64) {
    float S = 0.f, S2 = 0.f;
    for (int p = 0; p < 4; ++p) { S += red[0][p][tid]; S2 += red[1][p][tid]; }
    float mu = S * (1.f / 256.f);
    float var = fmaxf(S2 * (1.f / 256.f) - mu * mu, 0.f);
    stats[0][tid] = mu;
    stats[1][tid] = rsqrtf(var + 1e-5f);
  }
  __syncthreads();
  size_t base = ((size_t)b * NTOK) + n0;
  float mu = stats[0][tok], rs = stats[1][tok];
  for (int cc = 0; cc < 64; ++cc) {
    int c = cc * 4 + cl;
    float v = xb[(size_t)c * NTOK + tok];
    float y = (v - mu) * rs * g[c] + bta[c];
    t[(base + tok) * DIMC + c] = f2bf(y);
  }
}

// --------------------- shared 64x64 WMMA GEMM tile body --------------------
// 4 waves/block; wave w owns rows w*16..w*16+15 of the tile (4 accumulators).
// lds_a: [row][k] (pairs contiguous for A frags); lds_b: [n][k] (pairs
// contiguous for B frags).
__device__ __forceinline__ void gemm_tile_64x64(
    const u16* __restrict__ A, int lda, const u16* __restrict__ B, int ldb,
    int row0, int col0, int K, u16* lds_a, u16* lds_b, v8f* acc, int tid) {
  int wave = tid >> 5, lane = tid & 31, hi = lane >> 4, ln = lane & 15;
  for (int k0 = 0; k0 < K; k0 += 32) {
    __syncthreads();                       // prior tile fully consumed
    #pragma unroll
    for (int i = 0; i < 8; ++i) {          // 64x32 bf16 A tile as 1024 dwords
      int idx = i * 128 + tid;
      int r = idx >> 4, wc = idx & 15;
      ((u32*)lds_a)[idx] =
          *(const u32*)(A + (size_t)(row0 + r) * lda + k0 + wc * 2);
    }
    #pragma unroll
    for (int i = 0; i < 16; ++i) {         // B tile transposed into [n][k]
      int idx = i * 128 + tid;
      int kk = idx >> 6, nn = idx & 63;
      lds_b[nn * 32 + kk] = B[(size_t)(k0 + kk) * ldb + col0 + nn];
    }
    __syncthreads();
    ABfrag af;
    #pragma unroll
    for (int i = 0; i < 8; ++i)
      af.w[i] = *(const u32*)&lds_a[(wave * 16 + ln) * 32 + kpat(i, hi)];
    #pragma unroll
    for (int jb = 0; jb < 4; ++jb) {
      ABfrag bf;
      #pragma unroll
      for (int i = 0; i < 8; ++i)
        bf.w[i] = *(const u32*)&lds_b[(jb * 16 + ln) * 32 + kpat(i, hi)];
      acc[jb] = wmma_bf16(af.v, bf.v, acc[jb]);
    }
  }
}

// --------------------------- kernel 3: QKV GEMM ----------------------------
__global__ __launch_bounds__(128) void qkv_gemm_kernel(
    const u16* __restrict__ t, const u16* __restrict__ wq,
    u16* __restrict__ qb, u16* __restrict__ kb, u16* __restrict__ vb) {
  __shared__ u16 lds_a[64 * 32];
  __shared__ u16 lds_b[64 * 32];
  int tid = threadIdx.x, wave = tid >> 5, lane = tid & 31;
  int hi = lane >> 4, ln = lane & 15;
  int row0 = blockIdx.x * 64, col0 = blockIdx.y * 64;
  v8f acc[4] = {};
  gemm_tile_64x64(t, DIMC, wq, 3 * DIMC, row0, col0, DIMC, lds_a, lds_b, acc, tid);
  #pragma unroll
  for (int jb = 0; jb < 4; ++jb) {
    #pragma unroll
    for (int i = 0; i < 8; ++i) {
      int m = hi * 8 + i;
      int token = row0 + wave * 16 + m;
      int gc = col0 + jb * 16 + ln;          // (three, head, d)
      int three = gc >> 8, head = (gc >> 6) & 3, d = gc & 63;
      size_t bb = token >> 12, nn = token & 4095;
      u16* dst = (three == 0) ? qb : ((three == 1) ? kb : vb);
      dst[(((bb * NHEAD + head) << 12) + nn) * HDIM + d] = f2bf(acc[jb][i]);
    }
  }
}

// ------------------------ kernel 4: flash attention ------------------------
// grid = (N/64, B*h); block = 128 (4 waves). Each wave owns 16 query rows.
// Q and K tiles arrive via TDM async tensor loads; V is transposed into LDS.
__global__ __launch_bounds__(128) void flash_attn_kernel(
    const u16* __restrict__ qb, const u16* __restrict__ kb,
    const u16* __restrict__ vb, u16* __restrict__ ob) {
  __shared__ u16 lds_q[64 * 64];
  __shared__ u16 lds_k[64 * 64];
  __shared__ u16 lds_vt[64 * 64];            // [d][key]
  __shared__ u16 lds_p[4][16 * 64];          // per-wave P staging
  int qt = blockIdx.x, bh = blockIdx.y;
  int tid = threadIdx.x, wave = tid >> 5, lane = tid & 31;
  int hi = lane >> 4, ln = lane & 15;
  const u16* Q = qb + (size_t)bh * NTOK * HDIM + (size_t)qt * 64 * HDIM;
  const u16* K = kb + (size_t)bh * NTOK * HDIM;
  const u16* V = vb + (size_t)bh * NTOK * HDIM;

  if (wave == 0)
    tdm_load_bf16((u32)(size_t)&lds_q[0], (const void*)Q, 64 * HDIM);

  float m_old[8], l_sum[8];
  #pragma unroll
  for (int i = 0; i < 8; ++i) { m_old[i] = -3.0e38f; l_sum[i] = 0.f; }
  v8f o[4] = {};
  ABfrag aq[2];

  for (int kt = 0; kt < NTOK / 64; ++kt) {
    if (wave == 0)
      tdm_load_bf16((u32)(size_t)&lds_k[0], (const void*)(K + (size_t)kt * 64 * HDIM),
                    64 * HDIM);
    #pragma unroll
    for (int i = 0; i < 32; ++i) {           // V tile -> LDS transposed
      int idx = i * 128 + tid;
      int key = idx >> 6, d = idx & 63;
      lds_vt[d * 64 + key] = V[(size_t)kt * 64 * HDIM + idx];
    }
    if (wave == 0) __builtin_amdgcn_s_wait_tensorcnt(0);
    __syncthreads();
    if (kt == 0) {                           // Q fragments (persist in VGPRs)
      #pragma unroll
      for (int kc = 0; kc < 2; ++kc)
        #pragma unroll
        for (int i = 0; i < 8; ++i)
          aq[kc].w[i] =
              *(const u32*)&lds_q[(wave * 16 + ln) * 64 + kc * 32 + kpat(i, hi)];
    }
    // ---- S = Q K^T (scaled): 4 col-blocks x 2 K-steps ----
    v8f s[4];
    #pragma unroll
    for (int jb = 0; jb < 4; ++jb) {
      v8f sa = {};
      #pragma unroll
      for (int kc = 0; kc < 2; ++kc) {
        ABfrag bf;                           // B = K^T: lds_k row-major works
        #pragma unroll
        for (int i = 0; i < 8; ++i)
          bf.w[i] =
              *(const u32*)&lds_k[(jb * 16 + ln) * 64 + kc * 32 + kpat(i, hi)];
        sa = wmma_bf16(aq[kc].v, bf.v, sa);
      }
      s[jb] = sa * 0.125f;                   // 1/sqrt(64)
    }
    // ---- online softmax (rows live across 16-lane groups) ----
    #pragma unroll
    for (int i = 0; i < 8; ++i) {
      float tm = -3.0e38f;
      #pragma unroll
      for (int jb = 0; jb < 4; ++jb) tm = fmaxf(tm, s[jb][i]);
      #pragma unroll
      for (int off = 1; off < 16; off <<= 1)
        tm = fmaxf(tm, __shfl_xor(tm, off, 32));
      float mn = fmaxf(m_old[i], tm);
      float sc = __expf(m_old[i] - mn);
      m_old[i] = mn;
      l_sum[i] *= sc;
      #pragma unroll
      for (int db = 0; db < 4; ++db) o[db][i] *= sc;
      float rs = 0.f;
      #pragma unroll
      for (int jb = 0; jb < 4; ++jb) {
        float p = __expf(s[jb][i] - mn);
        s[jb][i] = p;
        rs += p;
      }
      #pragma unroll
      for (int off = 1; off < 16; off <<= 1) rs += __shfl_xor(rs, off, 32);
      l_sum[i] += rs;
    }
    // ---- P -> LDS (C-layout to A-layout relayout), then O += P V ----
    #pragma unroll
    for (int jb = 0; jb < 4; ++jb)
      #pragma unroll
      for (int i = 0; i < 8; ++i)
        lds_p[wave][(hi * 8 + i) * 64 + jb * 16 + ln] = f2bf(s[jb][i]);
    asm volatile("s_wait_dscnt 0" ::: "memory");  // wave-local store->load
    ABfrag pa[2];
    #pragma unroll
    for (int kc = 0; kc < 2; ++kc)
      #pragma unroll
      for (int i = 0; i < 8; ++i)
        pa[kc].w[i] = *(const u32*)&lds_p[wave][ln * 64 + kc * 32 + kpat(i, hi)];
    #pragma unroll
    for (int db = 0; db < 4; ++db) {
      #pragma unroll
      for (int kc = 0; kc < 2; ++kc) {
        ABfrag bf;                           // B = V: lds_vt [d][key]
        #pragma unroll
        for (int i = 0; i < 8; ++i)
          bf.w[i] =
              *(const u32*)&lds_vt[(db * 16 + ln) * 64 + kc * 32 + kpat(i, hi)];
        o[db] = wmma_bf16(pa[kc].v, bf.v, o[db]);
      }
    }
    __syncthreads();                         // tiles consumed; safe to overwrite
  }
  // ---- normalize and emit o[token][h*64+d] ----
  float inv_l[8];
  #pragma unroll
  for (int i = 0; i < 8; ++i) inv_l[i] = 1.f / l_sum[i];
  int b = bh >> 2, h = bh & 3;
  #pragma unroll
  for (int db = 0; db < 4; ++db)
    #pragma unroll
    for (int i = 0; i < 8; ++i) {
      int m = hi * 8 + i;
      size_t token = (size_t)b * NTOK + qt * 64 + wave * 16 + m;
      int c = h * HDIM + db * 16 + ln;
      ob[token * DIMC + c] = f2bf(o[db][i] * inv_l[i]);
    }
}

// --------------------------- kernel 5: out proj ----------------------------
__global__ __launch_bounds__(128) void proj_gemm_kernel(
    const u16* __restrict__ ob, const u16* __restrict__ wpT,
    const float* __restrict__ bias, float* __restrict__ out) {
  __shared__ u16 lds_a[64 * 32];
  __shared__ u16 lds_b[64 * 32];
  int tid = threadIdx.x, wave = tid >> 5, lane = tid & 31;
  int hi = lane >> 4, ln = lane & 15;
  int row0 = blockIdx.x * 64, col0 = blockIdx.y * 64;
  v8f acc[4] = {};
  gemm_tile_64x64(ob, DIMC, wpT, DIMC, row0, col0, DIMC, lds_a, lds_b, acc, tid);
  #pragma unroll
  for (int jb = 0; jb < 4; ++jb) {
    #pragma unroll
    for (int i = 0; i < 8; ++i) {
      int m = hi * 8 + i;
      int token = row0 + wave * 16 + m;
      int c = col0 + jb * 16 + ln;
      size_t b = token >> 12, n = token & 4095;
      out[((size_t)b * DIMC + c) * NTOK + n] = acc[jb][i] + bias[c];
    }
  }
}

// ------------------------------- launcher ----------------------------------
extern "C" void kernel_launch(void* const* d_in, const int* in_sizes, int n_in,
                              void* d_out, int out_size, void* d_ws, size_t ws_size,
                              hipStream_t stream) {
  const float* x      = (const float*)d_in[0];
  const float* ln_g   = (const float*)d_in[1];
  const float* ln_b   = (const float*)d_in[2];
  const float* w_qkv  = (const float*)d_in[3];
  const float* w_proj = (const float*)d_in[4];
  const float* b_proj = (const float*)d_in[5];
  float* out = (float*)d_out;

  // workspace layout (bf16 buffers), ~42.5 MB total
  char* ws = (char*)d_ws;
  const size_t SZ_T  = (size_t)16384 * DIMC * 2;      // 8 MB
  const size_t SZ_WQ = (size_t)DIMC * 3 * DIMC * 2;   // 384 KB
  const size_t SZ_WP = (size_t)DIMC * DIMC * 2;       // 128 KB
  const size_t SZ_H  = (size_t)4 * NHEAD * NTOK * HDIM * 2;  // 8 MB each
  size_t off = 0;
  u16* t_bf  = (u16*)(ws + off); off += SZ_T;
  u16* wq_bf = (u16*)(ws + off); off += SZ_WQ;
  u16* wp_bf = (u16*)(ws + off); off += SZ_WP;
  u16* qb    = (u16*)(ws + off); off += SZ_H;
  u16* kb    = (u16*)(ws + off); off += SZ_H;
  u16* vb    = (u16*)(ws + off); off += SZ_H;
  u16* ob    = (u16*)(ws + off); off += SZ_T;

  convert_weights_kernel<<<768, 256, 0, stream>>>(w_qkv, w_proj, wq_bf, wp_bf);
  ln_bf16_kernel<<<256, 256, 0, stream>>>(x, ln_g, ln_b, t_bf);
  qkv_gemm_kernel<<<dim3(256, 12), 128, 0, stream>>>(t_bf, wq_bf, qb, kb, vb);
  flash_attn_kernel<<<dim3(64, 16), 128, 0, stream>>>(qb, kb, vb, ob);
  proj_gemm_kernel<<<dim3(256, 4), 128, 0, stream>>>(ob, wp_bf, b_proj, out);
}
